// Graph_81174881894890
// MI455X (gfx1250) — matrix operand order repaired and checked by reading, not compiled, not run on hardware
//
#include <hip/hip_runtime.h>
#include <hip/hip_bf16.h>
#include <stdint.h>

// ---------------------------------------------------------------------------
// Types for CDNA5 WMMA (wave32): bf16 A/B fragments, f32 accumulator.
// ---------------------------------------------------------------------------
typedef __attribute__((ext_vector_type(16))) __bf16 v16bf;
typedef __attribute__((ext_vector_type(8)))  __bf16 bf16x8;   // 16 bytes
typedef __attribute__((ext_vector_type(8)))  float  v8f;
typedef __attribute__((ext_vector_type(4)))  float  f32x4;
typedef f32x4 __attribute__((aligned(4)))           f32x4u;   // align-4-ok b128 load

#define BM 64
#define BN 32
#define KBLK 64          // two 32-wide WMMA sub-chunks per barrier

enum { ACT_NONE = 0, ACT_ELU = 1, ACT_SIGMOID = 2 };

// Model constants (from the reference).
#define ND_  218
#define NC_  271
#define NN_  489      // ND + NC
#define DIN_ 512
#define FF_  1024
#define HH_  8
#define EE_  20000
#define ECC_ 8000

static inline int ceil_div_i(int a, int b) { return (a + b - 1) / b; }
static inline int64_t ceil_div_l(int64_t a, int64_t b) { return (a + b - 1) / b; }

// ---------------------------------------------------------------------------
// Generic batched GEMM with WMMA (bf16 MAC, f32 accumulate).
//   C[b] = act( A[b] @ B[b] (+ Cin[b]) (+ bias) )
// A: MxK (row major, lda), batch stride sA.
// B: KxN (row major, ldb) or, if TRANSB, NxK with row stride ldb.
// C: MxN (ldc), batch stride sC.
//
// Block = 256 threads = 8 waves -> 64x32 C tile; waves tile 4x2 of 16x16.
// LDS tiles are stored in WMMA *fragment order*: frag[lane][slot] so each
// operand fragment is one contiguous 32-byte read (2 x ds_load_b128) and
// each staged k-octet is one 16-byte ds_store_b128.
//
// A(16x32) LDS order -> element (r, c) of a 16x32 tile lives at
//   lane = (r&15) + 16*(c>>4),       slot = c&15
// B(32x16) LDS order (same k order as A) -> element (k, n) lives at
//   lane = (n&15) + 16*(k>>4),      slot = k&15          (k-contiguous!)
//
// FAST (compile-time): K % KBLK == 0 and (TRANSB or N % BN == 0).
// All guards are then pure index clamps + value selects -> zero divergent
// branches in the staging path (no exec-mask ladders).
// ---------------------------------------------------------------------------
template <bool TRANSB, bool FAST>
__global__ void gemm_wmma_kernel(
    const float* __restrict__ A, int64_t lda, int64_t sA,
    const float* __restrict__ B, int64_t ldb, int64_t sB,
    const float* __restrict__ bias,
    const float* __restrict__ Cin,
    float* __restrict__ C, int64_t ldc, int64_t sC,
    int M, int Nn, int K, int act)
{
    // [k-subchunk][wave row/col tile][lane][slot]
    __shared__ alignas(16) __bf16 As[2][4][32][16];   // 8 KB
    __shared__ alignas(16) __bf16 Bs[2][2][32][16];   // 4 KB

    const int batch = blockIdx.z;
    A += (int64_t)batch * sA;
    B += (int64_t)batch * sB;
    const float* CinB = Cin ? (Cin + (int64_t)batch * sC) : nullptr;
    C += (int64_t)batch * sC;

    const int tileM = blockIdx.y * BM;
    const int tileN = blockIdx.x * BN;
    const int tid   = threadIdx.x;
    const int wave  = tid >> 5;
    const int lane  = tid & 31;
    const int rt    = wave & 3;          // wave row tile (0..3) -> rows 16*rt
    const int nt    = wave >> 2;         // wave col tile (0..1) -> cols 16*nt
    const int hlf   = lane >> 4;
    const int l16   = lane & 15;

    v8f acc = {};

    for (int k0 = 0; k0 < K; k0 += KBLK) {
        // ---- Stage A tile: 64 rows x 64 k. Unit = (row, k-octet): 512 units,
        // exactly 2 per thread. Branchless: clamped indices + value selects.
#pragma unroll
        for (int uu = 0; uu < 2; ++uu) {
            int u = tid + uu * 256;
            int r = u >> 3;              // 0..63
            int o = u & 7;               // k-octet within 64
            int gr = tileM + r;
            bool rowok = (gr < M);
            int grc = rowok ? gr : (M - 1);
            int c = k0 + 8 * o;
            const float* srow = A + (int64_t)grc * lda;
            float v[8];
            if (FAST) {
                f32x4u p0 = *(const f32x4u*)(srow + c);
                f32x4u p1 = *(const f32x4u*)(srow + c + 4);
                v[0]=p0.x; v[1]=p0.y; v[2]=p0.z; v[3]=p0.w;
                v[4]=p1.x; v[5]=p1.y; v[6]=p1.z; v[7]=p1.w;
#pragma unroll
                for (int j = 0; j < 8; ++j) v[j] = rowok ? v[j] : 0.0f;
            } else {
#pragma unroll
                for (int j = 0; j < 8; ++j) {
                    int cj  = c + j;
                    int cjc = (cj < K) ? cj : (K - 1);     // clamped addr
                    float t = srow[cjc];                   // unconditional load
                    v[j] = (rowok && cj < K) ? t : 0.0f;   // value select
                }
            }
            bf16x8 bv;
#pragma unroll
            for (int j = 0; j < 8; ++j) bv[j] = (__bf16)v[j];
            int ks   = o >> 2;           // which 32-wide sub-chunk
            int oc   = o & 3;            // octet within sub-chunk
            int dl   = (r & 15) + 16 * (oc >> 1);
            int dr   = r >> 4;
            int dslt = (oc & 1) * 8;
            *(bf16x8*)&As[ks][dr][dl][dslt] = bv;
        }

        // ---- Stage B tile: 64 k x 32 n. 256 units, 1 per thread.
        if (TRANSB) {
            // B is Nn x K: k contiguous -> vector load, one b128 LDS store.
            int n = tid >> 3;            // 0..31
            int o = tid & 7;             // k-octet
            int gn = tileN + n;
            bool nok = (gn < Nn);
            int gnc = nok ? gn : (Nn - 1);
            int c = k0 + 8 * o;
            const float* srow = B + (int64_t)gnc * ldb;
            float v[8];
            if (FAST) {
                f32x4u p0 = *(const f32x4u*)(srow + c);
                f32x4u p1 = *(const f32x4u*)(srow + c + 4);
                v[0]=p0.x; v[1]=p0.y; v[2]=p0.z; v[3]=p0.w;
                v[4]=p1.x; v[5]=p1.y; v[6]=p1.z; v[7]=p1.w;
#pragma unroll
                for (int j = 0; j < 8; ++j) v[j] = nok ? v[j] : 0.0f;
            } else {
#pragma unroll
                for (int j = 0; j < 8; ++j) {
                    int cj  = c + j;
                    int cjc = (cj < K) ? cj : (K - 1);
                    float t = srow[cjc];
                    v[j] = (nok && cj < K) ? t : 0.0f;
                }
            }
            bf16x8 bv;
#pragma unroll
            for (int j = 0; j < 8; ++j) bv[j] = (__bf16)v[j];
            int ks   = o >> 2;
            int oc   = o & 3;                     // k32 = 8*oc + j
            int dl   = (n & 15) + 16 * (oc >> 1); // lane = n%16 + 16*(k32>>4)
            int dn   = n >> 4;
            int dslt = (oc & 1) * 8;              // slot = k32 & 15
            *(bf16x8*)&Bs[ks][dn][dl][dslt] = bv;
        } else {
            // B is K x Nn: n contiguous -> vector load a row octet, scatter
            // 8 bf16 to 8 lanes (same slot) in LDS.
            int k  = tid >> 2;           // 0..63
            int on = tid & 3;            // n-octet
            int gk = k0 + k;
            int nb = tileN + 8 * on;
            float v[8];
            if (FAST) {
                // K % KBLK == 0 -> gk < K always; Nn % BN == 0 -> full octet.
                const float* srow = B + (int64_t)gk * ldb;
                f32x4u p0 = *(const f32x4u*)(srow + nb);
                f32x4u p1 = *(const f32x4u*)(srow + nb + 4);
                v[0]=p0.x; v[1]=p0.y; v[2]=p0.z; v[3]=p0.w;
                v[4]=p1.x; v[5]=p1.y; v[6]=p1.z; v[7]=p1.w;
            } else {
                bool kok = (gk < K);
                int gkc = kok ? gk : (K - 1);
                const float* srow = B + (int64_t)gkc * ldb;
#pragma unroll
                for (int j = 0; j < 8; ++j) {
                    int nj  = nb + j;
                    int njc = (nj < Nn) ? nj : (Nn - 1);
                    float t = srow[njc];
                    v[j] = (kok && nj < Nn) ? t : 0.0f;
                }
            }
            int ks   = k >> 5;
            int k32  = k & 31;
            int dl16 = 16 * (k32 >> 4);
            int slot = k32 & 15;
#pragma unroll
            for (int j = 0; j < 8; ++j) {
                int n = 8 * on + j;
                Bs[ks][n >> 4][(n & 15) + dl16][slot] = (__bf16)v[j];
            }
        }

        __syncthreads();

        // ---- Two WMMAs per barrier; each operand is one 32B LDS read.
        {
            v16bf a0 = *(const v16bf*)&As[0][rt][lane][0];
            v16bf b0 = *(const v16bf*)&Bs[0][nt][lane][0];
            acc = __builtin_amdgcn_wmma_f32_16x16x32_bf16(
                      false, a0, false, b0, (short)0, acc, false, false);
            v16bf a1 = *(const v16bf*)&As[1][rt][lane][0];
            v16bf b1 = *(const v16bf*)&Bs[1][nt][lane][0];
            acc = __builtin_amdgcn_wmma_f32_16x16x32_bf16(
                      false, a1, false, b1, (short)0, acc, false, false);
        }
        __syncthreads();
    }

    // Epilogue: C/D layout -> VGPR r holds row (16*rt + r + hlf*8), col (16*nt + l16).
    const int gcol = tileN + 16 * nt + l16;
    if (gcol < Nn) {
#pragma unroll
        for (int r = 0; r < 8; ++r) {
            int grow = tileM + 16 * rt + r + hlf * 8;
            if (grow < M) {
                float v = acc[r];
                if (CinB) v += CinB[(int64_t)grow * ldc + gcol];
                if (bias) v += bias[gcol];
                if (act == ACT_ELU)          v = (v > 0.0f) ? v : (__expf(v) - 1.0f);
                else if (act == ACT_SIGMOID) v = 1.0f / (1.0f + __expf(-v));
                C[(int64_t)grow * ldc + gcol] = v;
            }
        }
    }
}

// ---------------------------------------------------------------------------
// es[n,h] = <h[n,h,:], a_s[h,:]>, ed likewise. One 256-thread block per row.
// ---------------------------------------------------------------------------
__global__ void head_dots_kernel(const float* __restrict__ h,
                                 const float* __restrict__ a_s,
                                 const float* __restrict__ a_d,
                                 float* __restrict__ es, float* __restrict__ ed,
                                 int H, int F)
{
    __shared__ float s1[256];
    __shared__ float s2[256];
    const int row  = blockIdx.x;       // n*H + head
    const int head = row % H;
    const float* hp = h   + (int64_t)row  * F;
    const float* ap = a_s + (int64_t)head * F;
    const float* dp = a_d + (int64_t)head * F;
    float x1 = 0.0f, x2 = 0.0f;
    for (int i = threadIdx.x; i < F; i += blockDim.x) {
        float v = hp[i];
        x1 += v * ap[i];
        x2 += v * dp[i];
    }
    s1[threadIdx.x] = x1;
    s2[threadIdx.x] = x2;
    __syncthreads();
    for (int s = 128; s > 0; s >>= 1) {
        if ((int)threadIdx.x < s) {
            s1[threadIdx.x] += s1[threadIdx.x + s];
            s2[threadIdx.x] += s2[threadIdx.x + s];
        }
        __syncthreads();
    }
    if (threadIdx.x == 0) { es[row] = s1[0]; ed[row] = s2[0]; }
}

// Monotonic int encoding for float atomic-max (handles negatives).
__device__ inline unsigned f2ord(float f) {
    unsigned u = __float_as_uint(f);
    return (u & 0x80000000u) ? ~u : (u | 0x80000000u);
}
__device__ inline float ord2f(unsigned u) {
    return (u & 0x80000000u) ? __uint_as_float(u & 0x7fffffffu)
                             : __uint_as_float(~u);
}

__global__ void edge_logits_kernel(const int* __restrict__ src,
                                   const int* __restrict__ dst,
                                   const float* __restrict__ es,
                                   const float* __restrict__ ed,
                                   float* __restrict__ e_out,
                                   unsigned* __restrict__ menc,
                                   int E, int H)
{
    int i = blockIdx.x * blockDim.x + threadIdx.x;
    if (i >= E * H) return;
    int e = i / H, h = i % H;
    int s = src[e], d = dst[e];
    float v = es[s * H + h] + ed[d * H + h];
    v = (v > 0.0f) ? v : 0.2f * v;           // leaky_relu(0.2)
    e_out[i] = v;
    atomicMax(&menc[d * H + h], f2ord(v));   // segment max over dst
}

__global__ void edge_weights_kernel(const int* __restrict__ dst,
                                    const float* __restrict__ e_in,
                                    const unsigned* __restrict__ menc,
                                    float* __restrict__ w_out,
                                    float* __restrict__ den,
                                    int E, int H)
{
    int i = blockIdx.x * blockDim.x + threadIdx.x;
    if (i >= E * H) return;
    int e = i / H, h = i % H;
    int d = dst[e];
    float m = ord2f(menc[d * H + h]);
    if (!isfinite(m)) m = 0.0f;              // matches jnp.where(isfinite(m), m, 0)
    float w = __expf(e_in[i] - m);
    w_out[i] = w;
    atomicAdd(&den[d * H + h], w);           // segment sum over dst
}

// agg[dst,h,:] += alpha * hfeat[src,h,:]; one block per (edge, head).
__global__ void edge_aggregate_kernel(const int* __restrict__ src,
                                      const int* __restrict__ dst,
                                      const float* __restrict__ w,
                                      const float* __restrict__ den,
                                      const float* __restrict__ hfeat,
                                      float* __restrict__ agg,
                                      int E, int H, int F)
{
    int eh = blockIdx.x;
    int e = eh / H, hd = eh % H;
    int s = src[e], d = dst[e];
    float alpha = w[eh] / (den[d * H + hd] + 1e-16f);
    const float* hp = hfeat + ((int64_t)s * H + hd) * F;
    float*       ap = agg   + ((int64_t)d * H + hd) * F;
    for (int i = threadIdx.x; i < F; i += blockDim.x)
        atomicAdd(&ap[i], alpha * hp[i]);
}

// out[n,f] = elu( mean_h agg[n,h,f] + bias[f] + res[n,f] )
__global__ void gat_finalize_kernel(const float* __restrict__ agg,
                                    const float* __restrict__ bias,
                                    const float* __restrict__ res,
                                    float* __restrict__ out,
                                    int Nn, int H, int F)
{
    int64_t i = (int64_t)blockIdx.x * blockDim.x + threadIdx.x;
    if (i >= (int64_t)Nn * F) return;
    int n = (int)(i / F), f = (int)(i % F);
    float s = 0.0f;
    for (int hh = 0; hh < H; ++hh)
        s += agg[((int64_t)n * H + hh) * F + f];
    s = s * (1.0f / H) + bias[f] + res[i];
    out[i] = (s > 0.0f) ? s : (__expf(s) - 1.0f);   // ELU
}

__global__ void fill_f32_kernel(float* p, float v, int64_t n) {
    int64_t i = (int64_t)blockIdx.x * blockDim.x + threadIdx.x;
    if (i < n) p[i] = v;
}
__global__ void fill_u32_kernel(unsigned* p, unsigned v, int64_t n) {
    int64_t i = (int64_t)blockIdx.x * blockDim.x + threadIdx.x;
    if (i < n) p[i] = v;
}

// ---------------------------------------------------------------------------
// Host orchestration
// ---------------------------------------------------------------------------
static void launch_gemm(hipStream_t stream,
                        const float* A, int64_t lda, int64_t sA,
                        const float* B, int64_t ldb, int64_t sB, int transB,
                        const float* bias, const float* Cin,
                        float* C, int64_t ldc, int64_t sC,
                        int M, int Nn, int K, int batch, int act)
{
    dim3 grid(ceil_div_i(Nn, BN), ceil_div_i(M, BM), batch);
    bool fast = (K % KBLK == 0) && (transB ? true : (Nn % BN == 0));
    if (transB) {
        if (fast)
            gemm_wmma_kernel<true, true><<<grid, 256, 0, stream>>>(
                A, lda, sA, B, ldb, sB, bias, Cin, C, ldc, sC, M, Nn, K, act);
        else
            gemm_wmma_kernel<true, false><<<grid, 256, 0, stream>>>(
                A, lda, sA, B, ldb, sB, bias, Cin, C, ldc, sC, M, Nn, K, act);
    } else {
        if (fast)
            gemm_wmma_kernel<false, true><<<grid, 256, 0, stream>>>(
                A, lda, sA, B, ldb, sB, bias, Cin, C, ldc, sC, M, Nn, K, act);
        else
            gemm_wmma_kernel<false, false><<<grid, 256, 0, stream>>>(
                A, lda, sA, B, ldb, sB, bias, Cin, C, ldc, sC, M, Nn, K, act);
    }
}

extern "C" void kernel_launch(void* const* d_in, const int* in_sizes, int n_in,
                              void* d_out, int out_size, void* d_ws, size_t ws_size,
                              hipStream_t stream)
{
    (void)in_sizes; (void)n_in; (void)out_size; (void)ws_size;

    const float* x     = (const float*)d_in[0];
    const int*   eidx  = (const int*)  d_in[1];
    const float* xcir  = (const float*)d_in[2];
    const int*   eidxc = (const int*)  d_in[3];
    const float* Wp    = (const float*)d_in[4];
    const float* bp    = (const float*)d_in[5];
    const float* Wpc   = (const float*)d_in[6];
    const float* bpc   = (const float*)d_in[7];
    const float* Wh0   = (const float*)d_in[8];
    const float* ash0  = (const float*)d_in[9];
    const float* adh0  = (const float*)d_in[10];
    const float* bh0   = (const float*)d_in[11];
    const float* rh0   = (const float*)d_in[12];
    const float* Wh1   = (const float*)d_in[13];
    const float* ash1  = (const float*)d_in[14];
    const float* adh1  = (const float*)d_in[15];
    const float* bh1   = (const float*)d_in[16];
    const float* Wc0   = (const float*)d_in[17];
    const float* asc0  = (const float*)d_in[18];
    const float* adc0  = (const float*)d_in[19];
    const float* bc0   = (const float*)d_in[20];
    const float* rc0   = (const float*)d_in[21];
    const float* Wc1   = (const float*)d_in[22];
    const float* asc1  = (const float*)d_in[23];
    const float* adc1  = (const float*)d_in[24];
    const float* bc1   = (const float*)d_in[25];
    const float* Wcnnh = (const float*)d_in[26];
    const float* bcnnh = (const float*)d_in[27];
    const float* Wcnnc = (const float*)d_in[28];
    const float* bcnnc = (const float*)d_in[29];
    const float* Wd    = (const float*)d_in[30];

    const int* srcH = eidx;
    const int* dstH = eidx + EE_;
    const int* srcC = eidxc;
    const int* dstC = eidxc + ECC_;

    // Workspace carve-out (256B aligned bump allocator).
    char* wp = (char*)d_ws;
    auto wsalloc = [&](size_t bytes) -> void* {
        void* p = (void*)wp;
        wp += (bytes + 255) & ~(size_t)255;
        return p;
    };
    float*    hbuf = (float*)   wsalloc((size_t)NN_ * HH_ * FF_ * 4);
    float*    agg  = (float*)   wsalloc((size_t)NN_ * HH_ * FF_ * 4);
    float*    esb  = (float*)   wsalloc((size_t)NN_ * HH_ * 4);
    float*    edb  = (float*)   wsalloc((size_t)NN_ * HH_ * 4);
    unsigned* menc = (unsigned*)wsalloc((size_t)NN_ * HH_ * 4);
    float*    den  = (float*)   wsalloc((size_t)NN_ * HH_ * 4);
    float*    ew   = (float*)   wsalloc((size_t)EE_ * HH_ * 4);
    float*    f0   = (float*)   wsalloc((size_t)NN_ * FF_ * 4);
    float*    f1   = (float*)   wsalloc((size_t)NN_ * FF_ * 4);
    float*    f2   = (float*)   wsalloc((size_t)NN_ * FF_ * 4);
    float*    resb = (float*)   wsalloc((size_t)NN_ * FF_ * 4);
    float*    zhet = (float*)   wsalloc((size_t)NN_ * FF_ * 4);
    float*    zcir = (float*)   wsalloc((size_t)NC_ * FF_ * 4);
    float*    t1   = (float*)   wsalloc((size_t)NN_ * FF_ * 4);

    // One full GAT layer (batched head GEMM + attention + finalize).
    auto gat_layer = [&](const float* xin, int Kin, int Nn,
                         const float* W, const float* a_s, const float* a_d,
                         const float* b, const float* rW,
                         const int* src, const int* dst, int Ecnt,
                         float* out) {
        // h[n,h,o] = x @ W[h]   (batched over heads; C batch stride = F)
        launch_gemm(stream, xin, Kin, 0,
                    W, FF_, (int64_t)Kin * FF_, 0,
                    nullptr, nullptr,
                    hbuf, (int64_t)HH_ * FF_, FF_,
                    Nn, FF_, Kin, HH_, ACT_NONE);
        head_dots_kernel<<<Nn * HH_, 256, 0, stream>>>(hbuf, a_s, a_d, esb, edb, HH_, FF_);

        int nh = Nn * HH_;
        fill_u32_kernel<<<ceil_div_i(nh, 256), 256, 0, stream>>>(menc, 0u, nh);
        fill_f32_kernel<<<ceil_div_i(nh, 256), 256, 0, stream>>>(den, 0.0f, nh);
        int64_t na = (int64_t)Nn * HH_ * FF_;
        fill_f32_kernel<<<(int)ceil_div_l(na, 256), 256, 0, stream>>>(agg, 0.0f, na);

        int eh = Ecnt * HH_;
        edge_logits_kernel<<<ceil_div_i(eh, 256), 256, 0, stream>>>(src, dst, esb, edb, ew, menc, Ecnt, HH_);
        edge_weights_kernel<<<ceil_div_i(eh, 256), 256, 0, stream>>>(dst, ew, menc, ew, den, Ecnt, HH_);
        edge_aggregate_kernel<<<eh, 256, 0, stream>>>(src, dst, ew, den, hbuf, agg, Ecnt, HH_, FF_);

        const float* resp;
        if (rW) {
            launch_gemm(stream, xin, Kin, 0, rW, FF_, 0, 0,
                        nullptr, nullptr, resb, FF_, 0,
                        Nn, FF_, Kin, 1, ACT_NONE);
            resp = resb;
        } else {
            resp = xin;   // residual = input (Kin == F in that case)
        }
        int64_t nf = (int64_t)Nn * FF_;
        gat_finalize_kernel<<<(int)ceil_div_l(nf, 256), 256, 0, stream>>>(agg, b, resp, out, Nn, HH_, FF_);
    };

    // Jumping-knowledge CNN: z[n,f] = sum_l feats_l[n,:] . Wc[f,l,:] + bc[f]
    auto jk = [&](const float* fA, const float* fB, const float* fC,
                  const float* Wc, const float* bc, float* z, int Nn) {
        launch_gemm(stream, fA, FF_, 0, Wc + 0 * FF_, 3 * FF_, 0, 1,
                    bc, nullptr, z, FF_, 0, Nn, FF_, FF_, 1, ACT_NONE);
        launch_gemm(stream, fB, FF_, 0, Wc + 1 * FF_, 3 * FF_, 0, 1,
                    nullptr, z, z, FF_, 0, Nn, FF_, FF_, 1, ACT_NONE);
        launch_gemm(stream, fC, FF_, 0, Wc + 2 * FF_, 3 * FF_, 0, 1,
                    nullptr, z, z, FF_, 0, Nn, FF_, FF_, 1, ACT_NONE);
    };

    // ---------------- Heterogeneous graph ----------------
    // feats0 = x @ Wp + bp
    launch_gemm(stream, x, DIN_, 0, Wp, FF_, 0, 0, bp, nullptr,
                f0, FF_, 0, NN_, FF_, DIN_, 1, ACT_NONE);
    gat_layer(x,  DIN_, NN_, Wh0, ash0, adh0, bh0, rh0,     srcH, dstH, EE_, f1);
    gat_layer(f1, FF_,  NN_, Wh1, ash1, adh1, bh1, nullptr, srcH, dstH, EE_, f2);
    jk(f0, f1, f2, Wcnnh, bcnnh, zhet, NN_);

    // ---------------- Circular graph ----------------
    launch_gemm(stream, xcir, NC_, 0, Wpc, FF_, 0, 0, bpc, nullptr,
                f0, FF_, 0, NC_, FF_, NC_, 1, ACT_NONE);
    gat_layer(xcir, NC_, NC_, Wc0, asc0, adc0, bc0, rc0,     srcC, dstC, ECC_, f1);
    gat_layer(f1,   FF_, NC_, Wc1, asc1, adc1, bc1, nullptr, srcC, dstC, ECC_, f2);
    jk(f0, f1, f2, Wcnnc, bcnnc, zcir, NC_);

    // ---------------- Final bilinear heads ----------------
    // t1 = z_het @ Wd   (489 x 1024)
    launch_gemm(stream, zhet, FF_, 0, Wd, FF_, 0, 0, nullptr, nullptr,
                t1, FF_, 0, NN_, FF_, FF_, 1, ACT_NONE);
    float* out = (float*)d_out;
    // ret   = sigmoid(t1[:218]  @ z_cir^T)  -> (218 x 271)
    launch_gemm(stream, t1, FF_, 0, zcir, FF_, 0, 1, nullptr, nullptr,
                out, NC_, 0, ND_, NC_, FF_, 1, ACT_SIGMOID);
    // ass_c = sigmoid(t1[218:] @ z_cir^T)  -> (271 x 271)
    launch_gemm(stream, t1 + (int64_t)ND_ * FF_, FF_, 0, zcir, FF_, 0, 1,
                nullptr, nullptr,
                out + (int64_t)ND_ * NC_, NC_, 0, NC_, NC_, FF_, 1, ACT_SIGMOID);
}
